// SANIrecursiveLayersModel_77764677861660
// MI455X (gfx1250) — matrix operand order repaired and checked by reading, not compiled, not run on hardware
//
#include <hip/hip_runtime.h>
#include <math.h>

#define BB 8
#define SS 128
#define HH 1024
#define EE 1024
#define VV 32000

typedef __bf16 bf16_t;
typedef __attribute__((ext_vector_type(16))) __bf16 bf16x16;
typedef __attribute__((ext_vector_type(4)))  __bf16 bf16x4;
typedef __attribute__((ext_vector_type(8)))  float  f32x8;
typedef __attribute__((ext_vector_type(4)))  float  f32x4;
typedef __attribute__((ext_vector_type(4)))  unsigned int u32x4;
typedef __attribute__((ext_vector_type(8)))  unsigned int u32x8;

// ---------------------------------------------------------------- WMMA helpers

__device__ __forceinline__ f32x8 wmma_bf16(bf16x16 a, bf16x16 b, f32x8 c) {
  return __builtin_amdgcn_wmma_f32_16x16x32_bf16(false, a, false, b, (short)0, c,
                                                 false, false);
}

// A fragment: 16x32 bf16, LDS row-major [16][ldk].
__device__ __forceinline__ bf16x16 frag_a(const bf16_t* sA, int ldk, int lane) {
  const int r = lane & 15;
  const int koff = (lane >> 4) << 3;
  const bf16_t* p = sA + r * ldk + koff;
  bf16x16 a;
#pragma unroll
  for (int e = 0; e < 8; ++e) a[e] = p[e];
#pragma unroll
  for (int e = 8; e < 16; ++e) a[e] = p[e + 8];
  return a;
}

// B fragment: 32x16 bf16, LDS stored transposed [16 cols][ldk K].
__device__ __forceinline__ bf16x16 frag_b(const bf16_t* sBt, int ldk, int lane) {
  const int c = lane & 15;
  const int koff = (lane >> 4) << 3;
  const bf16_t* p = sBt + c * ldk + koff;
  bf16x16 b;
#pragma unroll
  for (int e = 0; e < 8; ++e) b[e] = p[e];
#pragma unroll
  for (int e = 8; e < 16; ++e) b[e] = p[e + 8];
  return b;
}

// --------------------------------------------- CDNA5 data movers

__device__ __forceinline__ unsigned lds_addr_of(const void* p) {
  // LDS aperture: low 32 bits of the flat address are the wave-relative offset
  return (unsigned)(unsigned long long)(uintptr_t)p;
}

// TDM 2D tile load: global (bf16, row stride = strideElems) -> contiguous LDS.
// tileK elems contiguous per row, tileN rows. D# per CDNA5 ISA.
__device__ __forceinline__ void tdm_load_2d(const void* gsrc, unsigned ldsDst,
                                            unsigned tileK, unsigned tileN,
                                            unsigned strideElems,
                                            unsigned tdim0, unsigned tdim1) {
  const unsigned long long ga = (unsigned long long)(uintptr_t)gsrc;
  u32x4 g0;
  g0[0] = 1u;                                              // count=1, user D#
  g0[1] = ldsDst;                                          // lds_addr
  g0[2] = (unsigned)ga;                                    // global_addr[31:0]
  g0[3] = (unsigned)((ga >> 32) & 0x01ffffffull) | (2u << 30); // [56:32] | type=2
  u32x8 g1;
  g1[0] = 1u << 16;                                        // data_size=2B, mask=0
  g1[1] = (tdim0 & 0xffffu) << 16;                         // tensor_dim0[15:0]
  g1[2] = ((tdim0 >> 16) & 0xffffu) | ((tdim1 & 0xffffu) << 16);
  g1[3] = ((tdim1 >> 16) & 0xffffu) | (tileK << 16);       // tile_dim0
  g1[4] = tileN;                                           // tile_dim1, tile_dim2=0
  g1[5] = strideElems;                                     // tensor_dim0_stride lo32
  g1[6] = 0u;                                              // stride hi, dim1_stride
  g1[7] = 0u;
  asm volatile("tensor_load_to_lds %0, %1" : : "s"(g0), "s"(g1) : "memory");
}

__device__ __forceinline__ void tdm_wait() {
  __builtin_amdgcn_s_wait_tensorcnt(0);
}

// Per-lane async global->LDS 16-byte copy (ASYNCcnt path).
__device__ __forceinline__ void async_copy_b128(unsigned ldsDst, const void* gsrc) {
  asm volatile("global_load_async_to_lds_b128 %0, %1, off"
               : : "v"(ldsDst), "v"(gsrc) : "memory");
}
__device__ __forceinline__ void async_wait() {
  asm volatile("s_wait_asynccnt 0x0" : : : "memory");
}

// --------------------------------- K0: f32 [K][N] -> bf16 transposed [N][K]
__global__ void __launch_bounds__(256)
k_cvt_t(const float* __restrict__ W, bf16_t* __restrict__ Wt, int K, int N) {
  __shared__ float tile[32][33];
  const int kb = blockIdx.y * 32, nb = blockIdx.x * 32;
  const int tx = threadIdx.x & 31, ty = threadIdx.x >> 5;
  for (int r = ty; r < 32; r += 8)
    tile[r][tx] = W[(size_t)(kb + r) * N + nb + tx];
  __syncthreads();
  for (int r = ty; r < 32; r += 8)
    Wt[(size_t)(nb + r) * K + kb + tx] = (bf16_t)tile[tx][r];
}

// ------------------------------------------------- K1: embed + W_in + relu
__global__ void __launch_bounds__(128)
k_embed_in(const int* __restrict__ labels, const float* __restrict__ emb,
           const float* __restrict__ Win, const float* __restrict__ bin,
           float* __restrict__ h0) {
  __shared__ bf16_t sA[64 * 32];
  __shared__ bf16_t sB[64 * 32];
  const int tid = threadIdx.x;
  const int lane = tid & 31;
  const int w = tid >> 5;
  const int wm = w >> 1, wn = w & 1;
  const int mbase = blockIdx.y * 64;
  const int nbase = blockIdx.x * 64;
  f32x8 acc[2][2] = {};

  for (int ko = 0; ko < EE / 32; ++ko) {
    const int kb = ko * 32;
    for (int i = tid; i < 64 * 8; i += 128) {      // A: gathered embedding, f32x4
      const int r = i >> 3, kk = (i & 7) * 4;
      const int gr = mbase + r;
      const int b = gr & 7, t = gr >> 3;
      const int lab = labels[b * SS + t];
      f32x4 v = *(const f32x4*)&emb[(size_t)lab * EE + kb + kk];
      bf16x4 o; o[0]=(bf16_t)v[0]; o[1]=(bf16_t)v[1]; o[2]=(bf16_t)v[2]; o[3]=(bf16_t)v[3];
      *(bf16x4*)&sA[r * 32 + kk] = o;
    }
    for (int i = tid; i < (64 * 32) / 4; i += 128) { // B: W_in[k][n4] -> [n][k]
      const int k = i >> 4, n4 = (i & 15) * 4;
      f32x4 v = *(const f32x4*)&Win[(size_t)(kb + k) * HH + nbase + n4];
      sB[(n4 + 0) * 32 + k] = (bf16_t)v[0];
      sB[(n4 + 1) * 32 + k] = (bf16_t)v[1];
      sB[(n4 + 2) * 32 + k] = (bf16_t)v[2];
      sB[(n4 + 3) * 32 + k] = (bf16_t)v[3];
    }
    __syncthreads();
    bf16x16 a0 = frag_a(sA + (wm * 32) * 32, 32, lane);
    bf16x16 a1 = frag_a(sA + (wm * 32 + 16) * 32, 32, lane);
#pragma unroll
    for (int ni = 0; ni < 2; ++ni) {
      bf16x16 bfr = frag_b(sB + (wn * 32 + ni * 16) * 32, 32, lane);
      acc[0][ni] = wmma_bf16(a0, bfr, acc[0][ni]);
      acc[1][ni] = wmma_bf16(a1, bfr, acc[1][ni]);
    }
    __syncthreads();
  }
  const int c16 = lane & 15, hi = lane >> 4;
#pragma unroll
  for (int mi = 0; mi < 2; ++mi)
#pragma unroll
    for (int ni = 0; ni < 2; ++ni)
#pragma unroll
      for (int v = 0; v < 8; ++v) {
        const int row = mbase + wm * 32 + mi * 16 + hi * 8 + v;
        const int col = nbase + wn * 32 + ni * 16 + c16;
        h0[row * HH + col] = fmaxf(acc[mi][ni][v] + bin[col], 0.0f);
      }
}

// ------------------------------------------------- K2: one recursion step
// B tiles come from pre-transposed bf16 W_sani^T [n=1024][k=2048] via TDM.
__global__ void __launch_bounds__(256)
k_scan(const float* __restrict__ hin, float* __restrict__ hout,
       float* __restrict__ outS, const bf16_t* __restrict__ Wst,
       const float* __restrict__ bs,
       const float* __restrict__ g, const float* __restrict__ be, int m) {
  __shared__ bf16_t sB[512 * 32];      // 32 KB (one half of the cols)
  __shared__ bf16_t sA[16 * 32];       //  1 KB
  __shared__ float  red[2][16][128];   // 16 KB LN partials
  __shared__ float  stats[16][2];
  const int tid = threadIdx.x, lane = tid & 31, w = tid >> 5;
  const int rbase = (m + 1) * BB + blockIdx.x * 16;
  const int colbase = w * 128;
  const int half = (w >= 4);
  const unsigned sB_lds = lds_addr_of(sB);
  f32x8 acc[8] = {};

  for (int ko = 0; ko < (2 * HH) / 32; ++ko) {
    const int kb = ko * 32;
    if (tid < 128) {                                 // A: comb rows, f32x4 + cvt
      const int r = tid >> 3, kk = (tid & 7) * 4;
      const int gr = rbase + r;
      bf16x4 o = {};
      if (gr < SS * BB) {
        const int k = kb + kk;
        const float* src = (k < HH) ? &hin[(gr - BB) * HH + k]
                                    : &hin[gr * HH + (k - HH)];
        f32x4 v = *(const f32x4*)src;
        o[0]=(bf16_t)v[0]; o[1]=(bf16_t)v[1]; o[2]=(bf16_t)v[2]; o[3]=(bf16_t)v[3];
      }
      *(bf16x4*)&sA[r * 32 + kk] = o;
    }
    if (tid == 0)                                    // B cols 0..511 via TDM
      tdm_load_2d(Wst + kb, sB_lds, 32, 512, 2 * HH, 2 * HH, HH);
    tdm_wait();
    __syncthreads();
    if (!half) {
      bf16x16 af = frag_a(sA, 32, lane);
#pragma unroll
      for (int f = 0; f < 8; ++f)
        acc[f] = wmma_bf16(af, frag_b(sB + (colbase + f * 16) * 32, 32, lane), acc[f]);
    }
    __syncthreads();
    if (tid == 0)                                    // B cols 512..1023 via TDM
      tdm_load_2d(Wst + (size_t)512 * (2 * HH) + kb, sB_lds, 32, 512, 2 * HH, 2 * HH, HH);
    tdm_wait();
    __syncthreads();
    if (half) {
      bf16x16 af = frag_a(sA, 32, lane);
#pragma unroll
      for (int f = 0; f < 8; ++f)
        acc[f] = wmma_bf16(af, frag_b(sB + (colbase - 512 + f * 16) * 32, 32, lane), acc[f]);
    }
    __syncthreads();
  }

  // epilogue: val = relu(z + b)*0.1 + h ; then LayerNorm over H
  const int c16 = lane & 15, hi = lane >> 4;
  float vals[8][8];
#pragma unroll
  for (int f = 0; f < 8; ++f) {
    const int col = colbase + f * 16 + c16;
    const float bcol = bs[col];
#pragma unroll
    for (int v = 0; v < 8; ++v) {
      const int gr = rbase + hi * 8 + v;
      float x = fmaxf(acc[f][v] + bcol, 0.0f) * 0.1f;
      vals[f][v] = x + ((gr < SS * BB) ? hin[gr * HH + col] : 0.0f);
    }
  }
#pragma unroll
  for (int v = 0; v < 8; ++v) {
    float a = 0.f, b2 = 0.f;
#pragma unroll
    for (int f = 0; f < 8; ++f) { a += vals[f][v]; b2 += vals[f][v] * vals[f][v]; }
    const int row = hi * 8 + v;
    red[0][row][w * 16 + c16] = a;
    red[1][row][w * 16 + c16] = b2;
  }
  __syncthreads();
  {
    const int row = tid >> 4, j = tid & 15;
    float a = 0.f, b2 = 0.f;
#pragma unroll
    for (int q = 0; q < 8; ++q) { a += red[0][row][j * 8 + q]; b2 += red[1][row][j * 8 + q]; }
    __syncthreads();
    red[0][row][j] = a; red[1][row][j] = b2;
  }
  __syncthreads();
  if (tid < 16) {
    float a = 0.f, b2 = 0.f;
#pragma unroll
    for (int j = 0; j < 16; ++j) { a += red[0][tid][j]; b2 += red[1][tid][j]; }
    const float mean = a * (1.0f / HH);
    const float var = b2 * (1.0f / HH) - mean * mean;
    stats[tid][0] = mean;
    stats[tid][1] = rsqrtf(var + 1e-5f);
  }
  __syncthreads();
#pragma unroll
  for (int v = 0; v < 8; ++v) {
    const int row = hi * 8 + v;
    const int gr = rbase + row;
    if (gr < SS * BB) {
      const float mean = stats[row][0], rstd = stats[row][1];
      const int t = gr >> 3, b = gr & 7;
#pragma unroll
      for (int f = 0; f < 8; ++f) {
        const int col = colbase + f * 16 + c16;
        const float o = (vals[f][v] - mean) * rstd * g[col] + be[col];
        hout[gr * HH + col] = o;
        if (t == m + 1) outS[(b * SS + m) * HH + col] = o;
      }
    }
  }
}

// ------------------------------------------------- K3: dense -> gelu -> LN
__global__ void __launch_bounds__(256)
k_head(const float* __restrict__ xin, const bf16_t* __restrict__ Wdt,
       const float* __restrict__ bd, const float* __restrict__ g,
       const float* __restrict__ be, bf16_t* __restrict__ yb) {
  __shared__ bf16_t sB[512 * 32];
  __shared__ bf16_t sA[16 * 32];
  __shared__ float  red[2][16][128];
  __shared__ float  stats[16][2];
  const int tid = threadIdx.x, lane = tid & 31, w = tid >> 5;
  const int rbase = blockIdx.x * 16;
  const int colbase = w * 128;
  const int half = (w >= 4);
  const unsigned sB_lds = lds_addr_of(sB);
  f32x8 acc[8] = {};

  for (int ko = 0; ko < HH / 32; ++ko) {
    const int kb = ko * 32;
    if (tid < 128) {
      const int r = tid >> 3, kk = (tid & 7) * 4;
      f32x4 v = *(const f32x4*)&xin[(rbase + r) * HH + kb + kk];
      bf16x4 o; o[0]=(bf16_t)v[0]; o[1]=(bf16_t)v[1]; o[2]=(bf16_t)v[2]; o[3]=(bf16_t)v[3];
      *(bf16x4*)&sA[r * 32 + kk] = o;
    }
    if (tid == 0)
      tdm_load_2d(Wdt + kb, sB_lds, 32, 512, HH, HH, HH);
    tdm_wait();
    __syncthreads();
    if (!half) {
      bf16x16 af = frag_a(sA, 32, lane);
#pragma unroll
      for (int f = 0; f < 8; ++f)
        acc[f] = wmma_bf16(af, frag_b(sB + (colbase + f * 16) * 32, 32, lane), acc[f]);
    }
    __syncthreads();
    if (tid == 0)
      tdm_load_2d(Wdt + (size_t)512 * HH + kb, sB_lds, 32, 512, HH, HH, HH);
    tdm_wait();
    __syncthreads();
    if (half) {
      bf16x16 af = frag_a(sA, 32, lane);
#pragma unroll
      for (int f = 0; f < 8; ++f)
        acc[f] = wmma_bf16(af, frag_b(sB + (colbase - 512 + f * 16) * 32, 32, lane), acc[f]);
    }
    __syncthreads();
  }

  const int c16 = lane & 15, hi = lane >> 4;
  float vals[8][8];
#pragma unroll
  for (int f = 0; f < 8; ++f) {
    const int col = colbase + f * 16 + c16;
    const float bcol = bd[col];
#pragma unroll
    for (int v = 0; v < 8; ++v) {
      float x = acc[f][v] + bcol;
      vals[f][v] = 0.5f * x * (1.0f + erff(x * 0.70710678118654752f));
    }
  }
#pragma unroll
  for (int v = 0; v < 8; ++v) {
    float a = 0.f, b2 = 0.f;
#pragma unroll
    for (int f = 0; f < 8; ++f) { a += vals[f][v]; b2 += vals[f][v] * vals[f][v]; }
    const int row = hi * 8 + v;
    red[0][row][w * 16 + c16] = a;
    red[1][row][w * 16 + c16] = b2;
  }
  __syncthreads();
  {
    const int row = tid >> 4, j = tid & 15;
    float a = 0.f, b2 = 0.f;
#pragma unroll
    for (int q = 0; q < 8; ++q) { a += red[0][row][j * 8 + q]; b2 += red[1][row][j * 8 + q]; }
    __syncthreads();
    red[0][row][j] = a; red[1][row][j] = b2;
  }
  __syncthreads();
  if (tid < 16) {
    float a = 0.f, b2 = 0.f;
#pragma unroll
    for (int j = 0; j < 16; ++j) { a += red[0][tid][j]; b2 += red[1][tid][j]; }
    const float mean = a * (1.0f / HH);
    const float var = b2 * (1.0f / HH) - mean * mean;
    stats[tid][0] = mean;
    stats[tid][1] = rsqrtf(var + 1e-12f);
  }
  __syncthreads();
#pragma unroll
  for (int v = 0; v < 8; ++v) {
    const int row = hi * 8 + v;
    const float mean = stats[row][0], rstd = stats[row][1];
#pragma unroll
    for (int f = 0; f < 8; ++f) {
      const int col = colbase + f * 16 + c16;
      yb[(rbase + row) * HH + col] =
          (bf16_t)((vals[f][v] - mean) * rstd * g[col] + be[col]);
    }
  }
}

// ------------------------------------------------- K4: decoder GEMM -> logits
__global__ void __launch_bounds__(256)
k_decoder(const bf16_t* __restrict__ yb, const float* __restrict__ Wdec,
          const float* __restrict__ bdec, float* __restrict__ logits) {
  __shared__ bf16_t sA[32 * 32];    //  2 KB
  __shared__ bf16_t sB[256 * 32];   // 16 KB
  const int tid = threadIdx.x, lane = tid & 31, w = tid >> 5;
  const int wm = w >> 2, wn = w & 3;
  const int mbase = blockIdx.y * 32, nbase = blockIdx.x * 256;
  const unsigned sA_lds = lds_addr_of(sA);
  f32x8 acc[4] = {};

  for (int ko = 0; ko < HH / 32; ++ko) {
    const int kb = ko * 32;
    if (tid < 128) {                 // A: yb already bf16 -> async byte copy
      const int r = tid >> 2, c8 = (tid & 3) * 8;      // 8 bf16 = 16 B chunks
      async_copy_b128(sA_lds + (unsigned)(r * 32 + c8) * 2,
                      yb + (size_t)(mbase + r) * HH + kb + c8);
    }
    for (int i = tid; i < (256 * 32) / 4; i += 256) {  // B: W_dec f32x4 + cvt
      const int k = i >> 6, n4 = (i & 63) * 4;
      f32x4 v = *(const f32x4*)&Wdec[(size_t)(kb + k) * VV + nbase + n4];
      sB[(n4 + 0) * 32 + k] = (bf16_t)v[0];
      sB[(n4 + 1) * 32 + k] = (bf16_t)v[1];
      sB[(n4 + 2) * 32 + k] = (bf16_t)v[2];
      sB[(n4 + 3) * 32 + k] = (bf16_t)v[3];
    }
    async_wait();
    __syncthreads();
    bf16x16 af = frag_a(sA + wm * 16 * 32, 32, lane);
#pragma unroll
    for (int f = 0; f < 4; ++f)
      acc[f] = wmma_bf16(af, frag_b(sB + (wn * 64 + f * 16) * 32, 32, lane), acc[f]);
    __syncthreads();
  }
  const int c16 = lane & 15, hi = lane >> 4;
#pragma unroll
  for (int f = 0; f < 4; ++f) {
    const int col = nbase + wn * 64 + f * 16 + c16;
    const float bc = bdec[col];
#pragma unroll
    for (int v = 0; v < 8; ++v) {
      const int row = mbase + wm * 16 + hi * 8 + v;
      logits[(size_t)row * VV + col] = acc[f][v] + bc;
    }
  }
}

// ------------------------------------------------- K5: cross-entropy
__global__ void __launch_bounds__(256)
k_ce_row(const float* __restrict__ logits, const int* __restrict__ labels,
         float* __restrict__ rowloss) {
  __shared__ float sred[256];
  const int r = blockIdx.x, tid = threadIdx.x;
  const float* x = logits + (size_t)r * VV;
  float mx = -3.0e38f;
  for (int c = tid; c < VV; c += 256) mx = fmaxf(mx, x[c]);
  sred[tid] = mx; __syncthreads();
  for (int s = 128; s > 0; s >>= 1) {
    if (tid < s) sred[tid] = fmaxf(sred[tid], sred[tid + s]);
    __syncthreads();
  }
  mx = sred[0]; __syncthreads();
  float sum = 0.f;
  for (int c = tid; c < VV; c += 256) sum += expf(x[c] - mx);
  sred[tid] = sum; __syncthreads();
  for (int s = 128; s > 0; s >>= 1) {
    if (tid < s) sred[tid] += sred[tid + s];
    __syncthreads();
  }
  if (tid == 0) rowloss[r] = (mx + logf(sred[0])) - x[labels[r]];
}

__global__ void __launch_bounds__(256)
k_ce_final(const float* __restrict__ rowloss, float* __restrict__ out) {
  __shared__ float sred[256];
  float s = 0.f;
  for (int i = threadIdx.x; i < BB * SS; i += 256) s += rowloss[i];
  sred[threadIdx.x] = s; __syncthreads();
  for (int k = 128; k > 0; k >>= 1) {
    if (threadIdx.x < k) sred[threadIdx.x] += sred[threadIdx.x + k];
    __syncthreads();
  }
  if (threadIdx.x == 0) out[0] = sred[0] * (1.0f / (BB * SS));
}

// outputState rows for s in {126,127} are exact zeros in the reference
__global__ void k_zero_tail(float* __restrict__ outS) {
  const int i = blockIdx.x * blockDim.x + threadIdx.x;
  if (i < BB * 2 * HH) {
    const int b = i / (2 * HH);
    const int rem = i - b * (2 * HH);
    const int s = (SS - 2) + (rem >> 10);
    const int c = rem & (HH - 1);
    outS[(b * SS + s) * HH + c] = 0.0f;
  }
}

// ------------------------------------------------------------------- host
extern "C" void kernel_launch(void* const* d_in, const int* in_sizes, int n_in,
                              void* d_out, int out_size, void* d_ws, size_t ws_size,
                              hipStream_t stream) {
  const int*   labels = (const int*)  d_in[0];
  const float* emb    = (const float*)d_in[1];
  const float* Win    = (const float*)d_in[2];
  const float* bin    = (const float*)d_in[3];
  const float* Wsani  = (const float*)d_in[4];
  const float* bsani  = (const float*)d_in[5];
  const float* lng    = (const float*)d_in[6];
  const float* lnb    = (const float*)d_in[7];
  const float* Wdense = (const float*)d_in[8];
  const float* bdense = (const float*)d_in[9];
  const float* ln2g   = (const float*)d_in[10];
  const float* ln2b   = (const float*)d_in[11];
  const float* Wdec   = (const float*)d_in[12];
  const float* bdec   = (const float*)d_in[13];

  // workspace (~20.3 MB): h ping-pong, outputState, bf16 head act, bf16 W^T, losses
  char* ws = (char*)d_ws;
  float*  hA    = (float*)ws;  ws += (size_t)SS * BB * HH * sizeof(float);
  float*  hB    = (float*)ws;  ws += (size_t)SS * BB * HH * sizeof(float);
  float*  outS  = (float*)ws;  ws += (size_t)BB * SS * HH * sizeof(float);
  bf16_t* yb    = (bf16_t*)ws; ws += (size_t)BB * SS * HH * sizeof(bf16_t);
  bf16_t* Wst   = (bf16_t*)ws; ws += (size_t)2 * HH * HH * sizeof(bf16_t); // [H][2H]
  bf16_t* Wdt   = (bf16_t*)ws; ws += (size_t)HH * HH * sizeof(bf16_t);     // [H][H]
  float*  rloss = (float*)ws;

  float* loss   = (float*)d_out;
  float* logits = (float*)d_out + 1;   // [B,S,V]

  // pre-transpose + bf16-convert the recurrent / head weights (TDM sources)
  k_cvt_t<<<dim3(HH / 32, (2 * HH) / 32), 256, 0, stream>>>(Wsani, Wst, 2 * HH, HH);
  k_cvt_t<<<dim3(HH / 32, HH / 32), 256, 0, stream>>>(Wdense, Wdt, HH, HH);

  k_zero_tail<<<(BB * 2 * HH + 255) / 256, 256, 0, stream>>>(outS);
  k_embed_in<<<dim3(HH / 64, (SS * BB) / 64), 128, 0, stream>>>(labels, emb, Win, bin, hA);

  float* hin = hA;
  float* hout = hB;
  for (int m = 0; m <= SS - 3; ++m) {              // 126 sequential steps
    const int nrows = (SS - 1 - m) * BB;
    const int nblk = (nrows + 15) / 16;
    k_scan<<<nblk, 256, 0, stream>>>(hin, hout, outS, Wst, bsani, lng, lnb, m);
    float* t = hin; hin = hout; hout = t;
  }

  k_head<<<(BB * SS) / 16, 256, 0, stream>>>(outS, Wdt, bdense, ln2g, ln2b, yb);
  k_decoder<<<dim3(VV / 256, (BB * SS) / 32), 256, 0, stream>>>(yb, Wdec, bdec, logits);
  k_ce_row<<<BB * SS, 256, 0, stream>>>(logits, labels, rloss);
  k_ce_final<<<1, 256, 0, stream>>>(rloss, loss);
}
